// StyleGAN2Block_71871982731634
// MI455X (gfx1250) — matrix operand order repaired and checked by reading, not compiled, not run on hardware
//
#include <hip/hip_runtime.h>
#include <hip/hip_bf16.h>

// ---------------- problem constants (match reference) ----------------
#define BDIM 8
#define C    256      // CIN == COUT
#define HDIM 128
#define WDIM 128
#define LAT  512
#define HW   (HDIM*WDIM)

typedef __attribute__((ext_vector_type(16))) __bf16 v16bf;
typedef __attribute__((ext_vector_type(8)))  __bf16 v8bf;
typedef __attribute__((ext_vector_type(8)))  float  v8f;

// ---------------- LDS layout for the conv kernel ----------------
// B tile: 4 input rows x 132 padded cols x 32 ci  (bf16, ci contiguous)
//         (rows h0-1 .. h0+2 cover the 3x3 taps of output rows h0, h0+1)
// A tile: 9 taps x 128 cout x 32 ci               (bf16, ci contiguous)
#define BT_COLS  132
#define BT_ROWS  4
#define BT_ELEMS (BT_ROWS*BT_COLS*32)        // 16896
#define AT_ELEMS (9*128*32)                  // 36864
#define CONV_SMEM_BYTES ((BT_ELEMS + AT_ELEMS)*2)   // 107520 B < 160KB

// =====================================================================
// Kernel 1: styles (w @ A^T + b) and demod sigmas
// one block per batch element, 256 threads (one per channel)
// =====================================================================
__global__ __launch_bounds__(256)
void style_sigma_kernel(const float* __restrict__ wlat,
                        const float* __restrict__ A1w, const float* __restrict__ A1b,
                        const float* __restrict__ A2w, const float* __restrict__ A2b,
                        const float* __restrict__ A3w, const float* __restrict__ A3b,
                        const float* __restrict__ W1,  const float* __restrict__ W2,
                        float* __restrict__ s1, float* __restrict__ s2,
                        float* __restrict__ s3,
                        float* __restrict__ sg1, float* __restrict__ sg2) {
  __shared__ float wv[LAT];
  __shared__ float s1sq[C];
  __shared__ float s2sq[C];
  const int b = blockIdx.x;
  const int t = threadIdx.x;

  for (int i = t; i < LAT; i += 256) wv[i] = wlat[b*LAT + i];
  __syncthreads();

  float d1 = A1b[t], d2 = A2b[t], d3 = A3b[t];
  for (int i = 0; i < LAT; ++i) {
    const float wi = wv[i];
    d1 += wi * A1w[t*LAT + i];
    d2 += wi * A2w[t*LAT + i];
    d3 += wi * A3w[t*LAT + i];
  }
  s1[b*C + t] = d1;  s1sq[t] = d1*d1;
  s2[b*C + t] = d2;  s2sq[t] = d2*d2;
  s3[b*C + t] = d3;
  __syncthreads();

  // sigma[b,o] = sqrt( sum_i style^2[b,i] * sum_kl W[o,i,kl]^2 + eps )
  float a1 = 0.f, a2 = 0.f;
  for (int i = 0; i < C; ++i) {
    const float* p1 = W1 + ((size_t)t*C + i)*9;
    const float* p2 = W2 + ((size_t)t*C + i)*9;
    float q1 = 0.f, q2 = 0.f;
    #pragma unroll
    for (int k = 0; k < 9; ++k) { q1 += p1[k]*p1[k]; q2 += p2[k]*p2[k]; }
    a1 += s1sq[i]*q1;
    a2 += s2sq[i]*q2;
  }
  sg1[b*C + t] = sqrtf(a1 + 1e-8f);
  sg2[b*C + t] = sqrtf(a2 + 1e-8f);
}

// =====================================================================
// Kernel 2: modulated+demodulated 3x3 conv as bf16 WMMA implicit GEMM.
// grid = (2 cout-tiles, H/2, B); block = 512 threads (16 waves).
// Waves 0-7 produce output row h0, waves 8-15 row h0+1; both row-groups
// share one 128cout x 32ci x 9tap weight slab in LDS. Wave ws computes
// 16 cout x 128 px as 8 WMMA C tiles. K loop = 8 ci-chunks x 9 taps.
// Software pipeline: tap k+1's 18 ds_loads are issued BEFORE tap k's
// 8 WMMAs, fenced with sched_barrier so the scheduler cannot re-sink
// the loads next to their uses -> WMMA chain overlaps LDS latency.
// Epilogue fuses 1/sigma, noise, bias, leaky-relu(0.2).
// =====================================================================
__global__ __launch_bounds__(512)
void conv3x3_mod_wmma(const float* __restrict__ in,     // [B,C,H,W]
                      const float* __restrict__ style,  // [B,C]
                      const float* __restrict__ wgt,    // [C,C,3,3]
                      const float* __restrict__ sigma,  // [B,C]
                      const float* __restrict__ noise,  // [B,1,H,W]
                      const float* __restrict__ ns,     // [1]
                      const float* __restrict__ bias,   // [C]
                      float* __restrict__ out) {        // [B,C,H,W]
  extern __shared__ __align__(16) char smem_raw[];
  __bf16* Bt = (__bf16*)smem_raw;            // [4][132][32]
  __bf16* At = Bt + BT_ELEMS;                // [9][128][32]

  const int tid  = threadIdx.x;
  const int wid  = tid >> 5;
  const int lane = tid & 31;
  const int rg   = wid >> 3;                 // row group: 0 or 1
  const int ws8  = wid & 7;                  // wave within row group
  const int co0  = blockIdx.x * 128;
  const int h0   = blockIdx.y * 2;
  const int h    = h0 + rg;                  // this wave's output row
  const int b    = blockIdx.z;

  // WMMA operand lane mapping (16-bit A 16x32 / B 32x16, wave32):
  const int mrow   = lane & 15;   // A row (M) / B column (N) within tile
  const int kgrp   = lane >> 4;   // A: lanes<16 -> K0..15, lanes>=16 -> K16..31
  const int klo    = kgrp * 16;   // B: lanes<16 -> K0..15, lanes>=16 -> K16..31
  const int rowoff = kgrp * 8;    // C/D: lanes>=16 hold M rows 8..15

  v8f acc[8];
  {
    const v8f z = {0.f,0.f,0.f,0.f,0.f,0.f,0.f,0.f};
    #pragma unroll
    for (int i = 0; i < 8; ++i) acc[i] = z;
  }

  // fragment loader for one tap (2 A-loads + 16 B-loads, all ds_load_b128)
  auto load_tap = [&](int k9, v16bf& a, v16bf (&barr)[8]) {
    const int kh = k9 / 3;
    const int kw = k9 - kh*3;
    const __bf16* ap = At + ((k9*128 + ws8*16 + mrow)*32 + kgrp*16);
    const v8bf alo = *(const v8bf*)ap;
    const v8bf ahi = *(const v8bf*)(ap + 8);
    a = __builtin_shufflevector(alo, ahi,
        0,1,2,3,4,5,6,7,8,9,10,11,12,13,14,15);
    #pragma unroll
    for (int nt = 0; nt < 8; ++nt) {
      const int ncol = nt*16 + mrow;
      const __bf16* bp = Bt + (((kh + rg)*BT_COLS + ncol + kw)*32 + klo);
      const v8bf blo = *(const v8bf*)bp;
      const v8bf bhi = *(const v8bf*)(bp + 8);
      barr[nt] = __builtin_shufflevector(blo, bhi,
          0,1,2,3,4,5,6,7,8,9,10,11,12,13,14,15);
    }
  };

  for (int ci0 = 0; ci0 < C; ci0 += 32) {
    // ---- stage modulated input rows h0-1..h0+2 into LDS (bf16) ----
    for (int e = tid; e < BT_ELEMS; e += 512) {
      const int col = e % BT_COLS;           // fastest: coalesced global reads
      const int t2  = e / BT_COLS;
      const int ci  = t2 & 31;
      const int r   = t2 >> 5;               // 0..3
      const int sh  = h0 + r - 1;
      const int sw  = col - 1;
      float v = 0.f;
      if (sh >= 0 && sh < HDIM && sw >= 0 && sw < WDIM)
        v = in[(((size_t)b*C + ci0 + ci)*HDIM + sh)*WDIM + sw]
            * style[b*C + ci0 + ci];
      Bt[(r*BT_COLS + col)*32 + ci] = (__bf16)v;
    }
    // ---- stage weight slab [9 taps][128 cout][32 ci] into LDS ----
    for (int e = tid; e < AT_ELEMS; e += 512) {
      const int ci = e & 31;
      const int t2 = e >> 5;
      const int co = t2 & 127;
      const int k9 = t2 >> 7;
      const float v = wgt[(((size_t)(co0 + co))*C + ci0 + ci)*9 + k9];
      At[(k9*128 + co)*32 + ci] = (__bf16)v;
    }
    __syncthreads();

    // ---- 9 taps, double-buffered operand pipeline ----
    v16bf abuf[2];
    v16bf bbuf[2][8];
    load_tap(0, abuf[0], bbuf[0]);
    #pragma unroll
    for (int k9 = 0; k9 < 9; ++k9) {
      const int cur = k9 & 1;
      const int nxt = cur ^ 1;
      if (k9 < 8) load_tap(k9 + 1, abuf[nxt], bbuf[nxt]);
      __builtin_amdgcn_sched_barrier(0);   // keep next-tap loads above WMMAs
      #pragma unroll
      for (int nt = 0; nt < 8; ++nt) {
        acc[nt] = __builtin_amdgcn_wmma_f32_16x16x32_bf16(
            false, abuf[cur], false, bbuf[cur][nt], (short)0, acc[nt],
            false, false);
      }
      __builtin_amdgcn_sched_barrier(0);   // keep WMMAs below next loads
    }
    __syncthreads();
  }

  // ---- epilogue: demod, noise, bias, leaky-relu, store ----
  float rs[8], bi[8];
  #pragma unroll
  for (int r = 0; r < 8; ++r) {
    const int co = co0 + ws8*16 + rowoff + r;
    rs[r] = 1.0f / sigma[b*C + co];
    bi[r] = bias[co];
  }
  const float nsv = ns[0];
  #pragma unroll
  for (int nt = 0; nt < 8; ++nt) {
    const int wp = nt*16 + mrow;
    const float nz = nsv * noise[((size_t)b*HDIM + h)*WDIM + wp];
    #pragma unroll
    for (int r = 0; r < 8; ++r) {
      const int co = co0 + ws8*16 + rowoff + r;
      float v = acc[nt][r]*rs[r] + nz + bi[r];
      v = v > 0.f ? v : 0.2f*v;
      out[(((size_t)b*C + co)*HDIM + h)*WDIM + wp] = v;
    }
  }
}

// =====================================================================
// Kernel 3: modulated 1x1 conv 256 -> 3 (no demod). Memory bound.
// grid = (HW/256, B); 256 threads, one pixel each.
// =====================================================================
__global__ __launch_bounds__(256)
void rgb1x1_kernel(const float* __restrict__ fmap,   // [B,C,H,W] (= final out)
                   const float* __restrict__ s3,     // [B,C]
                   const float* __restrict__ W3,     // [3,C,1,1]
                   float* __restrict__ rgb) {        // [B,3,H,W]
  __shared__ float ws3[3*C];
  const int b = blockIdx.y;
  const int t = threadIdx.x;
  #pragma unroll
  for (int c = 0; c < 3; ++c)
    ws3[c*C + t] = W3[c*C + t] * s3[b*C + t];
  __syncthreads();

  const int p = blockIdx.x*256 + t;
  float a0 = 0.f, a1 = 0.f, a2 = 0.f;
  for (int ci = 0; ci < C; ++ci) {
    const float v = fmap[((size_t)b*C + ci)*HW + p];
    a0 += ws3[ci]       * v;
    a1 += ws3[C + ci]   * v;
    a2 += ws3[2*C + ci] * v;
  }
  rgb[((size_t)b*3 + 0)*HW + p] = a0;
  rgb[((size_t)b*3 + 1)*HW + p] = a1;
  rgb[((size_t)b*3 + 2)*HW + p] = a2;
}

// =====================================================================
// launch
// =====================================================================
extern "C" void kernel_launch(void* const* d_in, const int* in_sizes, int n_in,
                              void* d_out, int out_size, void* d_ws, size_t ws_size,
                              hipStream_t stream) {
  const float* fm     = (const float*)d_in[0];
  const float* wlat   = (const float*)d_in[1];
  const float* noise1 = (const float*)d_in[2];
  const float* noise2 = (const float*)d_in[3];
  const float* A1w    = (const float*)d_in[4];
  const float* A1b    = (const float*)d_in[5];
  const float* A2w    = (const float*)d_in[6];
  const float* A2b    = (const float*)d_in[7];
  const float* A3w    = (const float*)d_in[8];
  const float* A3b    = (const float*)d_in[9];
  const float* W1     = (const float*)d_in[10];
  const float* W2     = (const float*)d_in[11];
  const float* W3     = (const float*)d_in[12];
  const float* ns1    = (const float*)d_in[13];
  const float* ns2    = (const float*)d_in[14];
  const float* b1     = (const float*)d_in[15];
  const float* b2     = (const float*)d_in[16];

  // workspace layout (floats)
  float* wsf = (float*)d_ws;
  float* s1    = wsf;                 // B*C
  float* s2    = wsf + 1*BDIM*C;
  float* s3    = wsf + 2*BDIM*C;
  float* sg1   = wsf + 3*BDIM*C;
  float* sg2   = wsf + 4*BDIM*C;
  float* inter = wsf + 16384;         // [B,C,H,W] intermediate (64KB offset)

  float* out_fm = (float*)d_out;                       // [B,C,H,W]
  float* rgb    = out_fm + (size_t)BDIM*C*HW;          // [B,3,H,W]

  // 1) styles + sigmas
  style_sigma_kernel<<<dim3(BDIM), dim3(256), 0, stream>>>(
      wlat, A1w, A1b, A2w, A2b, A3w, A3b, W1, W2, s1, s2, s3, sg1, sg2);

  const dim3 cgrid(2, HDIM/2, BDIM);
  // 2) conv1: fm -> inter
  conv3x3_mod_wmma<<<cgrid, dim3(512), CONV_SMEM_BYTES, stream>>>(
      fm, s1, W1, sg1, noise1, ns1, b1, inter);
  // 3) conv2: inter -> out
  conv3x3_mod_wmma<<<cgrid, dim3(512), CONV_SMEM_BYTES, stream>>>(
      inter, s2, W2, sg2, noise2, ns2, b2, out_fm);
  // 4) 1x1 toRGB from final out
  rgb1x1_kernel<<<dim3(HW/256, BDIM), dim3(256), 0, stream>>>(
      out_fm, s3, W3, rgb);
}
